// Desimilar_block_71940702208422
// MI455X (gfx1250) — matrix operand order repaired and checked by reading, not compiled, not run on hardware
//
#include <hip/hip_runtime.h>

typedef unsigned short ushort_t;
typedef __attribute__((ext_vector_type(16))) __bf16 v16bf;
typedef __attribute__((ext_vector_type(8)))  float  v8f;

#define BATCH 4
#define CCH   64
#define LTOT  4096      // 64*64 positions per image
#define HIN   256
#define WIN_  256
#define KDOWN 1024      // 64 ci * 16 taps
#define KC1   1152      // 9 taps * 128 ci
#define BN_EPS 1e-5f

__device__ __forceinline__ unsigned short f2bf(float f) {
    unsigned int u = __float_as_uint(f);
    u += 0x7FFFu + ((u >> 16) & 1u);           // round-to-nearest-even
    return (unsigned short)(u >> 16);
}

union Frag16 { uint4 u[2]; v16bf v; };

// gfx1250 async global->LDS copy (ASYNCcnt-tracked, bypasses VGPRs).
// VDST operand carries the LDS byte offset = low 32 bits of the generic
// __shared__ pointer (flat aperture maps addr[31:0] -> LDS offset).
__device__ __forceinline__ void async_copy_b128(void* lds_dst, const void* gsrc) {
    unsigned lds_off = (unsigned)(unsigned long long)lds_dst;
    asm volatile("global_load_async_to_lds_b128 %0, %1, off"
                 :: "v"(lds_off), "v"(gsrc)
                 : "memory");
}
__device__ __forceinline__ void wait_async0() {
    asm volatile("s_wait_asynccnt 0x0" ::: "memory");
}

// ---------------------------------------------------------------------------
// Kernel 1: weight prep (fp32 -> bf16; w1 permuted to k = tap*128 + ci)
// ---------------------------------------------------------------------------
__global__ void k_prep(const float* __restrict__ wd, const float* __restrict__ w1,
                       ushort_t* __restrict__ wdbf, ushort_t* __restrict__ w1bf) {
    int idx = blockIdx.x * 256 + threadIdx.x;
    const int n1 = CCH * KDOWN;                 // wd is already [co][ci][tap] = [co][k]
    const int n2 = CCH * KC1;
    if (idx < n1) {
        wdbf[idx] = f2bf(wd[idx]);
    } else if (idx < n1 + n2) {
        int j  = idx - n1;
        int co = j / KC1, k = j % KC1;
        int tap = k >> 7, ci = k & 127;         // k = tap*128 + ci
        w1bf[j] = f2bf(w1[co * KC1 + ci * 9 + tap]);
    }
}

// ---------------------------------------------------------------------------
// Kernel 2: 4x4 stride-4 conv as implicit GEMM (M=16384,N=64,K=1024), bf16 WMMA
//           + bias + BN + ReLU; double-buffered LDS, async weight streaming.
//           Writes fp32 NHWC (xt) and bf16 concat ch 0..63.
// ---------------------------------------------------------------------------
__global__ void k_down_conv(const float* __restrict__ x,
                            const ushort_t* __restrict__ wdbf,
                            const float* __restrict__ bd,
                            const float* __restrict__ g,  const float* __restrict__ bb_,
                            const float* __restrict__ mu, const float* __restrict__ var,
                            float* __restrict__ xt, ushort_t* __restrict__ catb) {
    __shared__ alignas(16) ushort_t ldsA[2][64 * 32];   // 64 positions x 32 K
    __shared__ alignas(16) ushort_t ldsB[2][64 * 32];   // 64 N (co)     x 32 K
    const int t    = threadIdx.x;
    const int lane = t & 31, wv = t >> 5;
    const int hi   = lane >> 4;
    const int pos0 = blockIdx.x * 64;
    v8f acc[4] = {};

    auto loadA = [&](int kk, int buf) {                  // im2col gather (fp32->bf16)
        const int ci0 = kk >> 4;
#pragma unroll
        for (int i = 0; i < 16; ++i) {
            int e = t + (i << 7);
            int pos = e >> 5, kloc = e & 31;
            int ci  = ci0 + (kloc >> 4);
            int tap = kloc & 15;
            int p = pos0 + pos;
            int bidx = p >> 12, l = p & 4095;
            int oy = l >> 6, ox = l & 63;
            int iy = (oy << 2) + (tap >> 2), ix = (ox << 2) + (tap & 3);
            float v = x[((size_t)(bidx * CCH + ci) * HIN + iy) * WIN_ + ix];
            ldsA[buf][pos * 32 + kloc] = f2bf(v);
        }
    };
    auto loadB = [&](int kk, int buf) {                  // weights: async b128 stream
#pragma unroll
        for (int i = 0; i < 2; ++i) {
            int q = t + (i << 7);
            int n = q >> 2, koff = (q & 3) << 3;
            async_copy_b128(&ldsB[buf][n * 32 + koff], &wdbf[n * KDOWN + kk + koff]);
        }
    };

    loadB(0, 0);
    loadA(0, 0);
    wait_async0();
    __syncthreads();

    for (int kk = 0; kk < KDOWN; kk += 32) {
        const int cur = (kk >> 5) & 1;
        if (kk + 32 < KDOWN) {                           // prefetch next chunk
            loadB(kk + 32, cur ^ 1);
            loadA(kk + 32, cur ^ 1);
        }
        // A fragment (ISA 16-bit A 16x32 layout): row M = lane&15, K-half by lane>>4
        const uint4* rowA = (const uint4*)&ldsA[cur][(wv * 16 + (lane & 15)) * 32];
        Frag16 fa; fa.u[0] = rowA[hi]; fa.u[1] = rowA[2 + hi];
#pragma unroll
        for (int nt = 0; nt < 4; ++nt) {
            // B fragment: N = nt*16 + (lane&15); K-half by lane>>4 (same layout as A)
            const uint4* rowB = (const uint4*)&ldsB[cur][(nt * 16 + (lane & 15)) * 32];
            Frag16 fb; fb.u[0] = rowB[hi]; fb.u[1] = rowB[2 + hi];
            acc[nt] = __builtin_amdgcn_wmma_f32_16x16x32_bf16(
                false, fa.v, false, fb.v, (short)0, acc[nt], false, false);
        }
        wait_async0();
        __syncthreads();
    }
    // epilogue: D layout N = lane&15, M = r + 8*(lane>>4)
    const int n = lane & 15;
#pragma unroll
    for (int nt = 0; nt < 4; ++nt) {
        int co = nt * 16 + n;
        float scale = g[co] * rsqrtf(var[co] + BN_EPS);
#pragma unroll
        for (int r = 0; r < 8; ++r) {
            int p = pos0 + wv * 16 + r + 8 * hi;         // p = b*4096 + l (NHWC pixel)
            float val = acc[nt][r] + bd[co];
            val = (val - mu[co]) * scale + bb_[co];
            val = fmaxf(val, 0.0f);
            xt[(size_t)p * CCH + co]   = val;
            catb[(size_t)p * 128 + co] = f2bf(val);
        }
    }
}

// ---------------------------------------------------------------------------
// Kernel 3: 7x7-window dissimilarity, top-9 least-similar, mean-minus-self.
// One wave per position; writes bf16 into concat channels 64..127.
// ---------------------------------------------------------------------------
__global__ void k_feat(const float* __restrict__ xt, ushort_t* __restrict__ catb) {
    __shared__ float selfF[8][64];
    __shared__ float winv[8][64];
    __shared__ int   sel[8][16];
    const int wv = threadIdx.x >> 5, lane = threadIdx.x & 31;
    const int p = blockIdx.x * 8 + wv;
    const int b = p >> 12, l = p & 4095;
    const int row = l >> 6, col = l & 63;
    const float* base = xt + (size_t)b * LTOT * CCH;

    selfF[wv][lane]      = base[(size_t)l * CCH + lane];
    selfF[wv][lane + 32] = base[(size_t)l * CCH + lane + 32];
    __syncthreads();

    for (int j = lane; j < 64; j += 32) {
        float v = 1e20f;                              // matches pad value
        if (j < 49) {
            int r2 = row + j / 7 - 3, c2 = col + j % 7 - 3;
            if (r2 >= 0 && r2 < 64 && c2 >= 0 && c2 < 64) {
                const float* nb = base + (size_t)(r2 * 64 + c2) * CCH;
                float d2 = 0.f;
#pragma unroll
                for (int c = 0; c < CCH; ++c) {
                    float df = selfF[wv][c] - nb[c];
                    d2 += df * df;
                }
                float dist = sqrtf(fmaxf(d2, 1e-12f));
                v = 1.0f / (1.0f + dist);
            }
        }
        winv[wv][j] = v;
    }
    __syncthreads();
    if (lane == 0) {                                  // top-9 smallest sim, stable ties
        for (int k = 0; k < 9; ++k) {
            float best = 3.4e38f; int bi = 0;
            for (int j = 0; j < 49; ++j) {
                float vv = winv[wv][j];
                if (vv < best) { best = vv; bi = j; }
            }
            winv[wv][bi] = 3.4e38f;
            sel[wv][k] = bi;
        }
    }
    __syncthreads();
    for (int c = lane; c < CCH; c += 32) {
        float s = 0.f;
#pragma unroll
        for (int k = 0; k < 9; ++k) {
            int j = sel[wv][k];
            int r2 = row + j / 7 - 3, c2 = col + j % 7 - 3;
            s += base[(size_t)(r2 * 64 + c2) * CCH + c];
        }
        float o = s * (1.0f / 9.0f) - selfF[wv][c];
        catb[(size_t)p * 128 + 64 + c] = f2bf(o);
    }
}

// ---------------------------------------------------------------------------
// Kernel 4: 3x3 pad-1 conv over 128-ch concat (M=16384,N=64,K=1152) bf16 WMMA
//           + bias + BN + ReLU -> feat (NCHW fp32). Double-buffered LDS with
//           async-to-LDS streaming for both activations and weights.
// ---------------------------------------------------------------------------
__global__ void k_conv1(const ushort_t* __restrict__ catb,
                        const ushort_t* __restrict__ w1bf,
                        const float* __restrict__ b1,
                        const float* __restrict__ g,  const float* __restrict__ bb_,
                        const float* __restrict__ mu, const float* __restrict__ var,
                        float* __restrict__ feat) {
    __shared__ alignas(16) ushort_t ldsA[2][64 * 32];
    __shared__ alignas(16) ushort_t ldsB[2][64 * 32];
    const int t    = threadIdx.x;
    const int lane = t & 31, wv = t >> 5;
    const int hi   = lane >> 4;
    const int pos0 = blockIdx.x * 64;
    v8f acc[4] = {};

    auto loadA = [&](int kk, int buf) {              // im2col: async b128 per 8 channels
        const int tap = kk >> 7, ci0 = kk & 127;     // chunk lies within one tap
        const int dy = tap / 3 - 1, dx = tap % 3 - 1;
#pragma unroll
        for (int i = 0; i < 2; ++i) {
            int q = t + (i << 7);
            int pos = q >> 2, koff = (q & 3) << 3;
            int p = pos0 + pos;
            int bidx = p >> 12, l = p & 4095;
            int oy = l >> 6, ox = l & 63;
            int iy = oy + dy, ix = ox + dx;
            if (iy >= 0 && iy < 64 && ix >= 0 && ix < 64) {
                async_copy_b128(&ldsA[buf][pos * 32 + koff],
                                &catb[((size_t)(bidx * 4096 + iy * 64 + ix)) * 128
                                      + ci0 + koff]);
            } else {
                *(uint4*)&ldsA[buf][pos * 32 + koff] = make_uint4(0u, 0u, 0u, 0u);
            }
        }
    };
    auto loadB = [&](int kk, int buf) {
#pragma unroll
        for (int i = 0; i < 2; ++i) {
            int q = t + (i << 7);
            int n = q >> 2, koff = (q & 3) << 3;
            async_copy_b128(&ldsB[buf][n * 32 + koff], &w1bf[n * KC1 + kk + koff]);
        }
    };

    loadB(0, 0);
    loadA(0, 0);
    wait_async0();
    __syncthreads();

    for (int kk = 0; kk < KC1; kk += 32) {
        const int cur = (kk >> 5) & 1;
        if (kk + 32 < KC1) {
            loadB(kk + 32, cur ^ 1);
            loadA(kk + 32, cur ^ 1);
        }
        const uint4* rowA = (const uint4*)&ldsA[cur][(wv * 16 + (lane & 15)) * 32];
        Frag16 fa; fa.u[0] = rowA[hi]; fa.u[1] = rowA[2 + hi];
#pragma unroll
        for (int nt = 0; nt < 4; ++nt) {
            const uint4* rowB = (const uint4*)&ldsB[cur][(nt * 16 + (lane & 15)) * 32];
            Frag16 fb; fb.u[0] = rowB[hi]; fb.u[1] = rowB[2 + hi];
            acc[nt] = __builtin_amdgcn_wmma_f32_16x16x32_bf16(
                false, fa.v, false, fb.v, (short)0, acc[nt], false, false);
        }
        wait_async0();
        __syncthreads();
    }
    const int n = lane & 15;
#pragma unroll
    for (int nt = 0; nt < 4; ++nt) {
        int co = nt * 16 + n;
        float scale = g[co] * rsqrtf(var[co] + BN_EPS);
#pragma unroll
        for (int r = 0; r < 8; ++r) {
            int p = pos0 + wv * 16 + r + 8 * hi;
            int bidx = p >> 12, l = p & 4095;
            float val = acc[nt][r] + b1[co];
            val = (val - mu[co]) * scale + bb_[co];
            val = fmaxf(val, 0.0f);
            feat[((size_t)(bidx * CCH + co) << 12) + l] = val;
        }
    }
}

// ---------------------------------------------------------------------------
// Kernel 5: spatial mean of feat per (b, c)
// ---------------------------------------------------------------------------
__global__ void k_mean(const float* __restrict__ feat, float* __restrict__ attin) {
    __shared__ float red[256];
    const int bc = blockIdx.x;
    float s = 0.f;
    for (int i = threadIdx.x; i < LTOT; i += 256)
        s += feat[((size_t)bc << 12) + i];
    red[threadIdx.x] = s;
    __syncthreads();
    for (int st = 128; st > 0; st >>= 1) {
        if (threadIdx.x < st) red[threadIdx.x] += red[threadIdx.x + st];
        __syncthreads();
    }
    if (threadIdx.x == 0) attin[bc] = red[0] * (1.0f / (float)LTOT);
}

// ---------------------------------------------------------------------------
// Kernel 6: 1x1 conv (64x64 matvec per batch) + BN + sigmoid -> att
// ---------------------------------------------------------------------------
__global__ void k_att(const float* __restrict__ attin, const float* __restrict__ wa,
                      const float* __restrict__ g,  const float* __restrict__ bb_,
                      const float* __restrict__ mu, const float* __restrict__ var,
                      float* __restrict__ att) {
    const int b = blockIdx.x, co = threadIdx.x;
    float s = 0.f;
#pragma unroll 8
    for (int ci = 0; ci < CCH; ++ci)
        s += wa[co * CCH + ci] * attin[b * CCH + ci];
    float val = (s - mu[co]) * (g[co] * rsqrtf(var[co] + BN_EPS)) + bb_[co];
    att[b * CCH + co] = 1.0f / (1.0f + expf(-val));
}

// ---------------------------------------------------------------------------
// Kernel 7: out = bilinear_x4(feat) * att   (half-pixel centers, edge clamp)
// ---------------------------------------------------------------------------
__global__ void k_upsample(const float* __restrict__ feat, const float* __restrict__ att,
                           float* __restrict__ out) {
    const int idx = blockIdx.x * 256 + threadIdx.x;     // 4*64*256*256 threads
    const int ox = idx & 255;
    const int oy = (idx >> 8) & 255;
    const int c  = (idx >> 16) & 63;
    const int b  = idx >> 22;
    float fy = oy * 0.25f - 0.375f;
    float fx = ox * 0.25f - 0.375f;
    int y0 = (int)floorf(fy), x0 = (int)floorf(fx);
    float wy = fy - (float)y0, wx = fx - (float)x0;
    int y1 = min(max(y0 + 1, 0), 63), x1 = min(max(x0 + 1, 0), 63);
    y0 = min(max(y0, 0), 63);        x0 = min(max(x0, 0), 63);
    const float* f = feat + ((size_t)(b * CCH + c) << 12);
    float v = (1.f - wy) * ((1.f - wx) * f[y0 * 64 + x0] + wx * f[y0 * 64 + x1])
            +        wy  * ((1.f - wx) * f[y1 * 64 + x0] + wx * f[y1 * 64 + x1]);
    out[idx] = v * att[b * CCH + c];
}

// ---------------------------------------------------------------------------
extern "C" void kernel_launch(void* const* d_in, const int* in_sizes, int n_in,
                              void* d_out, int out_size, void* d_ws, size_t ws_size,
                              hipStream_t stream) {
    (void)in_sizes; (void)n_in; (void)out_size; (void)ws_size;
    const float* x     = (const float*)d_in[0];
    const float* wd    = (const float*)d_in[1];
    const float* bd    = (const float*)d_in[2];
    const float* bnd_g = (const float*)d_in[3];
    const float* bnd_b = (const float*)d_in[4];
    const float* bnd_m = (const float*)d_in[5];
    const float* bnd_v = (const float*)d_in[6];
    const float* w1    = (const float*)d_in[7];
    const float* b1    = (const float*)d_in[8];
    const float* bn1_g = (const float*)d_in[9];
    const float* bn1_b = (const float*)d_in[10];
    const float* bn1_m = (const float*)d_in[11];
    const float* bn1_v = (const float*)d_in[12];
    const float* wa    = (const float*)d_in[13];
    const float* bna_g = (const float*)d_in[14];
    const float* bna_b = (const float*)d_in[15];
    const float* bna_m = (const float*)d_in[16];
    const float* bna_v = (const float*)d_in[17];

    char* ws = (char*)d_ws;
    size_t o = 0;
    float*    xt    = (float*)(ws + o);    o += (size_t)BATCH * LTOT * CCH * 4;  // 4 MiB
    ushort_t* catb  = (ushort_t*)(ws + o); o += (size_t)BATCH * LTOT * 128 * 2;  // 4 MiB
    ushort_t* wdbf  = (ushort_t*)(ws + o); o += (size_t)CCH * KDOWN * 2;         // 128 KiB
    ushort_t* w1bf  = (ushort_t*)(ws + o); o += (size_t)CCH * KC1 * 2;           // 144 KiB
    float*    feat  = (float*)(ws + o);    o += (size_t)BATCH * CCH * LTOT * 4;  // 4 MiB
    float*    attin = (float*)(ws + o);    o += 256 * 4;
    float*    att   = (float*)(ws + o);    o += 256 * 4;

    k_prep     <<<(CCH * KDOWN + CCH * KC1 + 255) / 256, 256, 0, stream>>>(wd, w1, wdbf, w1bf);
    k_down_conv<<<256,   128, 0, stream>>>(x, wdbf, bd, bnd_g, bnd_b, bnd_m, bnd_v, xt, catb);
    k_feat     <<<2048,  256, 0, stream>>>(xt, catb);
    k_conv1    <<<256,   128, 0, stream>>>(catb, w1bf, b1, bn1_g, bn1_b, bn1_m, bn1_v, feat);
    k_mean     <<<BATCH * CCH, 256, 0, stream>>>(feat, attin);
    k_att      <<<BATCH, 64,  0, stream>>>(attin, wa, bna_g, bna_b, bna_m, bna_v, att);
    k_upsample <<<(BATCH * CCH * 256 * 256) / 256, 256, 0, stream>>>(feat, att, (float*)d_out);
}
